// Knowledge_Base_45260365365546
// MI455X (gfx1250) — matrix operand order repaired and checked by reading, not compiled, not run on hardware
//
#include <hip/hip_runtime.h>
#include <hip/hip_bf16.h>

typedef __attribute__((ext_vector_type(16))) _Float16 v16h;
typedef __attribute__((ext_vector_type(8)))  _Float16 v8h;
typedef __attribute__((ext_vector_type(8)))  float    v8f;

// Problem constants (fixed by the reference: B=4, S=2048, E=768, H=512, O=32)
constexpr int kM = 8192;            // B*S rows
constexpr int kE = 768;             // K of GEMM1
constexpr int kH = 512;             // N of GEMM1 / K of GEMM2
constexpr int kO = 32;              // codebook size
constexpr int kWaves = 8;
constexpr int kThreads = kWaves * 32;        // 256 threads = 8 wave32
constexpr int kRowsPerBlock = kWaves * 16;   // 128 rows / block
constexpr int kBlocks = kM / kRowsPerBlock;  // 64 blocks
constexpr int kJB = 8;                       // H 16-col tiles kept in regs per pass
constexpr int kCols = kJB * 16;              // 128 H-columns per pass
constexpr int kPasses = (kH / 16) / kJB;     // 4 passes
constexpr int kPitch = kCols + 8;            // LDS row pitch in halves (128 + 8 pad)
constexpr float kThresh = 0.3f;

__device__ __forceinline__ v16h cvt_v16h(float4 a0, float4 a1, float4 a2, float4 a3) {
  v16h r;
  r[0]  = (_Float16)a0.x; r[1]  = (_Float16)a0.y; r[2]  = (_Float16)a0.z; r[3]  = (_Float16)a0.w;
  r[4]  = (_Float16)a1.x; r[5]  = (_Float16)a1.y; r[6]  = (_Float16)a1.z; r[7]  = (_Float16)a1.w;
  r[8]  = (_Float16)a2.x; r[9]  = (_Float16)a2.y; r[10] = (_Float16)a2.z; r[11] = (_Float16)a2.w;
  r[12] = (_Float16)a3.x; r[13] = (_Float16)a3.y; r[14] = (_Float16)a3.z; r[15] = (_Float16)a3.w;
  return r;
}

__device__ __forceinline__ v16h cat_v16h(v8h c0, v8h c1) {
  v16h r;
#pragma unroll
  for (int i = 0; i < 8; ++i) { r[i] = c0[i]; r[8 + i] = c1[i]; }
  return r;
}

// ---- one-shot f32 -> f16 staging kernel (8 elements / thread) ----
__global__ __launch_bounds__(256)
void cvt_f32_to_f16(const float* __restrict__ src, _Float16* __restrict__ dst, int n) {
  const int i = (blockIdx.x * 256 + threadIdx.x) * 8;
  if (i + 8 <= n) {
    float4 a = *reinterpret_cast<const float4*>(src + i);
    float4 b = *reinterpret_cast<const float4*>(src + i + 4);
    v8h o;
    o[0] = (_Float16)a.x; o[1] = (_Float16)a.y; o[2] = (_Float16)a.z; o[3] = (_Float16)a.w;
    o[4] = (_Float16)b.x; o[5] = (_Float16)b.y; o[6] = (_Float16)b.z; o[7] = (_Float16)b.w;
    *reinterpret_cast<v8h*>(dst + i) = o;
  }
}

template <bool STAGED>
__global__ __launch_bounds__(kThreads)
void vq_softmax_fused(const float* __restrict__ A,      // [8192, 768] word_output
                      const float* __restrict__ W,      // [512, 768]
                      const float* __restrict__ bias,   // [512]
                      const float* __restrict__ opemb,  // [32, 512]
                      const _Float16* __restrict__ Ah,  // staged f16 copies (or null)
                      const _Float16* __restrict__ Wh,
                      const _Float16* __restrict__ Eh,
                      float* __restrict__ out)          // [8192, 32]
{
  __shared__ __align__(16) _Float16 tile[kWaves][16][kPitch]; // per-wave proj tile (f16)
  __shared__ float opn[kO];                                   // ||op_emb||^2

  const int tid  = threadIdx.x;
  const int wave = tid >> 5;
  const int lane = tid & 31;
  const int lm   = lane & 15;   // row (A) / col (B,C) within 16x16 tile
  const int lh   = lane >> 4;   // half-wave selector

  // Codebook norms (one thread per op; tiny, L2-hot; fp32 source for accuracy)
  if (tid < kO) {
    const float* er = opemb + tid * kH;
    float s = 0.f;
    for (int h = 0; h < kH; h += 4) {
      float4 v = *reinterpret_cast<const float4*>(er + h);
      s = fmaf(v.x, v.x, s); s = fmaf(v.y, v.y, s);
      s = fmaf(v.z, v.z, s); s = fmaf(v.w, v.w, s);
    }
    opn[tid] = s;
  }
  __syncthreads();

  const int mbase = blockIdx.x * kRowsPerBlock + wave * 16;
  const float*    arow  = A  + (size_t)(mbase + lm) * kE;
  const _Float16* arowh = STAGED ? (Ah + (size_t)(mbase + lm) * kE) : nullptr;

  v8f dotlo = {0,0,0,0,0,0,0,0};  // proj . op_emb[0..15]
  v8f dothi = {0,0,0,0,0,0,0,0};  // proj . op_emb[16..31]
  v8f nrm   = {0,0,0,0,0,0,0,0};  // partial ||proj row||^2 (this lane's columns)

  for (int pass = 0; pass < kPasses; ++pass) {
    const int hbase = pass * kCols;

    v8f acc[kJB];
#pragma unroll
    for (int jj = 0; jj < kJB; ++jj) acc[jj] = (v8f){0,0,0,0,0,0,0,0};

    // ---- GEMM1: proj tile (16 x 128) over K = 768, f16 WMMA, fp32 acc ----
    for (int kb = 0; kb < kE; kb += 32) {
      // A fragment 16x32 f16: lane = row lm, K = kb+16*lh+{0..15}
      v16h af;
      if constexpr (STAGED) {
        const _Float16* ap = arowh + kb + lh * 16;
        af = cat_v16h(*reinterpret_cast<const v8h*>(ap),
                      *reinterpret_cast<const v8h*>(ap + 8));
      } else {
        const float* ap = arow + kb + lh * 16;
        af = cvt_v16h(*reinterpret_cast<const float4*>(ap),
                      *reinterpret_cast<const float4*>(ap + 4),
                      *reinterpret_cast<const float4*>(ap + 8),
                      *reinterpret_cast<const float4*>(ap + 12));
      }
#pragma unroll
      for (int jj = 0; jj < kJB; ++jj) {
        // B fragment 32x16 f16: col n = lm -> W row (hbase+jj*16+lm), K = kb+16*lh+{0..15}
        v16h bf;
        if constexpr (STAGED) {
          const _Float16* wp = Wh + (size_t)(hbase + jj * 16 + lm) * kE + kb + lh * 16;
          bf = *reinterpret_cast<const v16h*>(wp);   // 32B aligned contiguous K-run
        } else {
          const float* wp = W + (size_t)(hbase + jj * 16 + lm) * kE + kb + lh * 16;
          bf = cvt_v16h(*reinterpret_cast<const float4*>(wp),
                        *reinterpret_cast<const float4*>(wp + 4),
                        *reinterpret_cast<const float4*>(wp + 8),
                        *reinterpret_cast<const float4*>(wp + 12));
        }
        acc[jj] = __builtin_amdgcn_wmma_f32_16x16x32_f16(
            false, af, false, bf, (short)0, acc[jj], false, false);
      }
    }

    // ---- bias add, norm accumulation, stage proj tile to LDS as f16 ----
#pragma unroll
    for (int jj = 0; jj < kJB; ++jj) {
      const float bv = bias[hbase + jj * 16 + lm];
#pragma unroll
      for (int r = 0; r < 8; ++r) {
        // C layout: element (row = r + 8*lh, col = lm) of this 16x16 tile
        float v = acc[jj][r] + bv;
        nrm[r] = fmaf(v, v, nrm[r]);
        tile[wave][r + 8 * lh][jj * 16 + lm] = (_Float16)v;
      }
    }
    // Tile is wave-private; DS ops are in-order per wave. Fence the compiler
    // and drain DScnt before re-reading as A-fragments.
    asm volatile("s_wait_dscnt 0" ::: "memory");

    // ---- GEMM2: dot(proj, op_emb) over this pass's 128 H-columns ----
#pragma unroll
    for (int kb = 0; kb < kCols; kb += 32) {
      const _Float16* lp = &tile[wave][lm][kb + lh * 16];
      v16h af2 = cat_v16h(*reinterpret_cast<const v8h*>(lp),
                          *reinterpret_cast<const v8h*>(lp + 8));

      // B: col n = op index; K = hbase+kb+16*lh+{0..15} contiguous in op_emb row
      v16h bl, bh;
      if constexpr (STAGED) {
        const _Float16* elo = Eh + (size_t)lm * kH + hbase + kb + lh * 16;
        bl = *reinterpret_cast<const v16h*>(elo);
        bh = *reinterpret_cast<const v16h*>(elo + 16 * kH);
      } else {
        const float* elo = opemb + (size_t)lm * kH + hbase + kb + lh * 16;
        const float* ehi = elo + 16 * kH;
        bl = cvt_v16h(*reinterpret_cast<const float4*>(elo),
                      *reinterpret_cast<const float4*>(elo + 4),
                      *reinterpret_cast<const float4*>(elo + 8),
                      *reinterpret_cast<const float4*>(elo + 12));
        bh = cvt_v16h(*reinterpret_cast<const float4*>(ehi),
                      *reinterpret_cast<const float4*>(ehi + 4),
                      *reinterpret_cast<const float4*>(ehi + 8),
                      *reinterpret_cast<const float4*>(ehi + 12));
      }
      dotlo = __builtin_amdgcn_wmma_f32_16x16x32_f16(
          false, af2, false, bl, (short)0, dotlo, false, false);
      dothi = __builtin_amdgcn_wmma_f32_16x16x32_f16(
          false, af2, false, bh, (short)0, dothi, false, false);
    }
    asm volatile("" ::: "memory");
  }

  // ---- distances, softmax over O=32, threshold, store ----
  const float onlo = opn[lm];
  const float onhi = opn[16 + lm];

#pragma unroll
  for (int r = 0; r < 8; ++r) {
    // Row norm: reduce over the 16 lanes of this half-wave (masks<16 stay in-half)
    float s = nrm[r];
    s += __shfl_xor(s, 1, 32);
    s += __shfl_xor(s, 2, 32);
    s += __shfl_xor(s, 4, 32);
    s += __shfl_xor(s, 8, 32);

    float dl = -sqrtf(fmaxf(fmaf(-2.f, dotlo[r], s) + onlo, 0.f));
    float dh = -sqrtf(fmaxf(fmaf(-2.f, dothi[r], s) + onhi, 0.f));

    float mx = fmaxf(dl, dh);
    mx = fmaxf(mx, __shfl_xor(mx, 1, 32));
    mx = fmaxf(mx, __shfl_xor(mx, 2, 32));
    mx = fmaxf(mx, __shfl_xor(mx, 4, 32));
    mx = fmaxf(mx, __shfl_xor(mx, 8, 32));

    float el = __expf(dl - mx);
    float eh = __expf(dh - mx);
    float den = el + eh;
    den += __shfl_xor(den, 1, 32);
    den += __shfl_xor(den, 2, 32);
    den += __shfl_xor(den, 4, 32);
    den += __shfl_xor(den, 8, 32);

    const float inv = 1.0f / den;
    float pl = el * inv;
    float ph = eh * inv;
    pl = (pl < kThresh) ? 0.f : pl;
    ph = (ph < kThresh) ? 0.f : ph;

    const int row = mbase + r + 8 * lh;   // matches C-tile row ownership
    out[row * kO + lm]      = pl;
    out[row * kO + 16 + lm] = ph;
  }
}

extern "C" void kernel_launch(void* const* d_in, const int* in_sizes, int n_in,
                              void* d_out, int out_size, void* d_ws, size_t ws_size,
                              hipStream_t stream) {
  (void)in_sizes; (void)n_in; (void)out_size;
  const float* A     = (const float*)d_in[0];  // word_output [4,2048,768]
  const float* W     = (const float*)d_in[1];  // W [512,768]
  const float* bias  = (const float*)d_in[2];  // b [512]
  const float* opemb = (const float*)d_in[3];  // op_emb [32,512]
  float* out = (float*)d_out;                  // prob [4,2048,32]

  constexpr size_t nA = (size_t)kM * kE;   // 6291456
  constexpr size_t nW = (size_t)kH * kE;   //  393216
  constexpr size_t nE = (size_t)kO * kH;   //   16384
  constexpr size_t needed = (nA + nW + nE) * sizeof(_Float16);

  if (ws_size >= needed) {
    _Float16* Ah = (_Float16*)d_ws;
    _Float16* Wh = Ah + nA;
    _Float16* Eh = Wh + nW;
    cvt_f32_to_f16<<<(int)(nA / 2048), 256, 0, stream>>>(A, Ah, (int)nA);
    cvt_f32_to_f16<<<(int)(nW / 2048), 256, 0, stream>>>(W, Wh, (int)nW);
    cvt_f32_to_f16<<<(int)(nE / 2048), 256, 0, stream>>>(opemb, Eh, (int)nE);
    vq_softmax_fused<true><<<kBlocks, kThreads, 0, stream>>>(
        A, W, bias, opemb, Ah, Wh, Eh, out);
  } else {
    vq_softmax_fused<false><<<kBlocks, kThreads, 0, stream>>>(
        A, W, bias, opemb, nullptr, nullptr, nullptr, out);
  }
}
